// BasicAttention_86809878987112
// MI455X (gfx1250) — matrix operand order repaired and checked by reading, not compiled, not run on hardware
//
#include <hip/hip_runtime.h>

#define B_       16
#define ENC_T    512
#define DEC_T    1024
#define D_       512
#define KW       31
#define PAD_     15
#define KPAD     64        // 62 (2*31) padded to 64 for the WMMA K loop
#define NTHREADS 512
#define NWAVES   16

typedef __attribute__((ext_vector_type(16))) __bf16 bf16x16;
typedef __attribute__((ext_vector_type(8)))  float  f32x8;

union FragU { uint4 q[2]; bf16x16 v; };

__device__ __forceinline__ unsigned short f2bf(float f) {
  union { float f; unsigned u; } x; x.f = f;
  unsigned r = x.u + 0x7FFFu + ((x.u >> 16) & 1u);   // round-to-nearest-even
  return (unsigned short)(r >> 16);
}

// tanh(x) = 1 - 2/(1 + e^{2x});  single v_exp_f32 + v_rcp_f32, no IEEE div chain
__device__ __forceinline__ float fast_tanh(float x) {
  x = fminf(15.0f, fmaxf(-15.0f, x));
  float e = __expf(2.0f * x);
  return 1.0f - 2.0f * __builtin_amdgcn_rcpf(1.0f + e);
}

__global__ __launch_bounds__(NTHREADS, 1)
void attn_scan_kernel(const float* __restrict__ enc,     // [B, ENC_T, D]
                      const float* __restrict__ mel,     // [B, DEC_T, D]
                      const int*   __restrict__ out_len, // [B]
                      const float* __restrict__ w_lin,   // [1, D]
                      const float* __restrict__ w_conv,  // [D, 2, KW]
                      float* __restrict__ out_ctx,       // [B, DEC_T, D]
                      float* __restrict__ out_score)     // [B, DEC_T, ENC_T]
{
  // LDS-resident recurrent state + operands (~142 KB of the 320 KB/WGP)
  __shared__ unsigned short U_s[ENC_T][KPAD];   // windowed [score|cum], bf16  (64 KB)
  __shared__ unsigned short Wt_s[D_][KPAD];     // conv weights, [d][j], bf16  (64 KB)
  __shared__ float  score_s[ENC_T];
  __shared__ float  cum_s[ENC_T];
  __shared__ float  ctx_s[D_];
  __shared__ float2 aw_s[D_];                   // {mel_t*mask + ctx, w_lin}: one ds_load_b64
  __shared__ float  logits_s[ENC_T];
  __shared__ float  wred_s[NWAVES];

  const int  b    = blockIdx.x;
  const int  tid  = threadIdx.x;
  const int  lane = tid & 31;
  const int  wid  = tid >> 5;
  const bool hi   = lane >= 16;
  const int  lrow = lane & 15;
  const int  olen = out_len[b];

  // ---------------- one-time init ----------------
  score_s[tid] = 0.f; cum_s[tid] = 0.f; ctx_s[tid] = 0.f;
  const float wl_v = w_lin[tid];                // thread `tid` owns d = tid
  U_s[tid][62] = 0; U_s[tid][63] = 0;           // K padding, constant
  Wt_s[tid][62] = 0; Wt_s[tid][63] = 0;
  for (int idx = tid; idx < D_ * 62; idx += NTHREADS) {
    int d = idx / 62, j = idx - d * 62;
    int c = (j < KW) ? 0 : 1;
    int k = (j < KW) ? j : (j - KW);
    Wt_s[d][j] = f2bf(w_conv[(d * 2 + c) * KW + k]);
  }
  __syncthreads();

  const float* encB = enc + (size_t)b * ENC_T * D_;

  for (int step = 0; step < DEC_T; ++step) {
    const float msk = (step < olen) ? 1.f : 0.f;

    // ---- phase A: aw = {ctx + masked mel_t, wl} ; rebuild window matrix U ----
    {
      float2 aw;
      aw.x = ctx_s[tid] + msk * mel[((size_t)b * DEC_T + step) * D_ + tid];
      aw.y = wl_v;
      aw_s[tid] = aw;
    }
    {
      const int t = tid;
      #pragma unroll 1
      for (int j = 0; j < KW; ++j) {
        int s = t - PAD_ + j;
        bool in = (s >= 0) && (s < ENC_T);
        U_s[t][j]      = f2bf(in ? score_s[s] : 0.f);
        U_s[t][KW + j] = f2bf(in ? cum_s[s]   : 0.f);
      }
    }
    __syncthreads();

    // ---- phase B: conv-GEMM via bf16 WMMA, fused tanh + w_lin reduce ----
    for (int ttl = 0; ttl < 2; ++ttl) {
      const int tt = wid * 2 + ttl;               // this wave's t-tile
      const int ta = tt * 16 + lrow;              // A row (t) for this lane
      const int c0 = hi ? 8 : 0;                  // A K-interleave per ISA layout
      const uint4* rowU = (const uint4*)(&U_s[ta][0]);  // 8 uint4 per row
      FragU a0, a1;
      a0.q[0] = rowU[(c0) >> 3];       a0.q[1] = rowU[(c0 + 16) >> 3];
      a1.q[0] = rowU[(32 + c0) >> 3];  a1.q[1] = rowU[(32 + c0 + 16) >> 3];

      float part[8];
      #pragma unroll
      for (int r = 0; r < 8; ++r) part[r] = 0.f;

      for (int dt = 0; dt < (D_ / 16); ++dt) {
        const int nb = dt * 16 + lrow;            // B col (d) for this lane
        const uint4* rowW = (const uint4*)(&Wt_s[nb][0]);
        const int kb0 = (hi ? 8 : 0) >> 3;        // B K-interleave per ISA layout
        FragU b0, b1;
        b0.q[0] = rowW[kb0];      b0.q[1] = rowW[kb0 + 2];
        b1.q[0] = rowW[kb0 + 4];  b1.q[1] = rowW[kb0 + 6];

        f32x8 acc = {0.f, 0.f, 0.f, 0.f, 0.f, 0.f, 0.f, 0.f};
        acc = __builtin_amdgcn_wmma_f32_16x16x32_bf16(false, a0.v, false, b0.v,
                                                      (short)0, acc, false, false);
        acc = __builtin_amdgcn_wmma_f32_16x16x32_bf16(false, a1.v, false, b1.v,
                                                      (short)0, acc, false, false);

        const float2 aw = aw_s[nb];               // one ds_load_b64
        #pragma unroll
        for (int r = 0; r < 8; ++r)
          part[r] += aw.y * fast_tanh(acc[r] + aw.x); // trans co-executes with XDL
      }
      // reduce the d dimension across the 16 lanes of each half-wave
      #pragma unroll
      for (int r = 0; r < 8; ++r) {
        float v = part[r];
        v += __shfl_xor(v, 1, 32);
        v += __shfl_xor(v, 2, 32);
        v += __shfl_xor(v, 4, 32);
        v += __shfl_xor(v, 8, 32);
        part[r] = v;
      }
      if (lrow == 0) {
        const int tbase = tt * 16 + (hi ? 8 : 0); // C layout: M = r (+8 for hi half)
        #pragma unroll
        for (int r = 0; r < 8; ++r) logits_s[tbase + r] = part[r];
      }
    }
    __syncthreads();

    // ---- phase C: softmax over enc_T, update cum, emit score ----
    {
      float l = logits_s[tid];
      float v = l;
      #pragma unroll
      for (int m = 16; m > 0; m >>= 1) v = fmaxf(v, __shfl_xor(v, m, 32));
      if (lane == 0) wred_s[wid] = v;
      __syncthreads();
      float mx = wred_s[0];
      #pragma unroll
      for (int i = 1; i < NWAVES; ++i) mx = fmaxf(mx, wred_s[i]);
      __syncthreads();                           // before wred_s reuse
      float e  = __expf(l - mx);
      float sv = e;
      #pragma unroll
      for (int m = 16; m > 0; m >>= 1) sv += __shfl_xor(sv, m, 32);
      if (lane == 0) wred_s[wid] = sv;
      __syncthreads();
      float tot = 0.f;
      #pragma unroll
      for (int i = 0; i < NWAVES; ++i) tot += wred_s[i];
      float sc = e * __builtin_amdgcn_rcpf(tot); // tot >= e > 0, well conditioned
      score_s[tid] = sc;
      cum_s[tid]  += sc;
      out_score[((size_t)b * DEC_T + step) * ENC_T + tid] = sc;  // unmasked, per ref
    }
    __syncthreads();

    // ---- phase D: new ctx = score @ enc (L2-resident streaming matvec) ----
    {
      const int d = tid;
      float acc = 0.f;
      const float* p = encB + d;
      #pragma unroll 4
      for (int t = 0; t < ENC_T; ++t)
        acc += score_s[t] * p[(size_t)t * D_];   // LDS-broadcast * coalesced
      ctx_s[d] = acc;
      out_ctx[((size_t)b * DEC_T + step) * D_ + d] = acc * msk;  // dec_mask applied
    }
    __syncthreads();
  }
}

extern "C" void kernel_launch(void* const* d_in, const int* in_sizes, int n_in,
                              void* d_out, int out_size, void* d_ws, size_t ws_size,
                              hipStream_t stream) {
  const float* enc    = (const float*)d_in[0];  // encoder_outputs [B,ENC_T,D]
  const float* mel    = (const float*)d_in[1];  // melenc_outputs  [B,DEC_T,D]
  // d_in[2] = encoder_lengths (unused by the math)
  const int*   olen   = (const int*)d_in[3];    // output_lengths  [B]
  const float* w_lin  = (const float*)d_in[4];  // [1,D]
  const float* w_conv = (const float*)d_in[5];  // [D,2,KW]

  float* out       = (float*)d_out;
  float* out_ctx   = out;                               // att_outs   [B,DEC_T,D]
  float* out_score = out + (size_t)B_ * DEC_T * D_;     // att_scores [B,DEC_T,ENC_T]

  attn_scan_kernel<<<dim3(B_), dim3(NTHREADS), 0, stream>>>(
      enc, mel, olen, w_lin, w_conv, out_ctx, out_score);
}
